// Net_67293547593971
// MI455X (gfx1250) — matrix operand order repaired and checked by reading, not compiled, not run on hardware
//
#include <hip/hip_runtime.h>

typedef __attribute__((ext_vector_type(16))) _Float16 v16h;
typedef __attribute__((ext_vector_type(8)))  _Float16 v8h;
typedef __attribute__((ext_vector_type(8)))  float    v8f;
typedef _Float16 h16;

// A-fragment loader: lane holds k = {half*8..+7, 16+half*8..+7} -> two b128s
__device__ __forceinline__ v16h load_afrag(const h16* __restrict__ p) {
  v8h lo = *(const v8h*)p;
  v8h hi = *(const v8h*)(p + 16);
  v16h r;
#pragma unroll
  for (int j = 0; j < 8; ++j) { r[j] = lo[j]; r[8 + j] = hi[j]; }
  return r;
}

// ------------------------------------------------------------------
// f32 -> f16 copy
// ------------------------------------------------------------------
__global__ void f32_to_f16_kernel(const float* __restrict__ s, h16* __restrict__ d, size_t n) {
  size_t i  = (size_t)blockIdx.x * blockDim.x + threadIdx.x;
  size_t st = (size_t)gridDim.x * blockDim.x;
  for (; i < n; i += st) d[i] = (h16)s[i];
}

// Pack weight W[KxN] f32 -> fragment-order f16: out[kt][n][kk], zero padded.
__global__ void pack_b_kernel(const float* __restrict__ W, int K, int N, int Npad,
                              h16* __restrict__ out, size_t total) {
  size_t t  = (size_t)blockIdx.x * blockDim.x + threadIdx.x;
  size_t st = (size_t)gridDim.x * blockDim.x;
  for (; t < total; t += st) {
    size_t kt  = t / ((size_t)Npad * 32);
    size_t rem = t - kt * (size_t)Npad * 32;
    int n  = (int)(rem >> 5);
    int kk = (int)(rem & 31);
    int k  = (int)(kt * 32 + (((kk & 8) << 1) | ((kk & 16) >> 1) | (kk & 7)));
    out[t] = (k < K && n < N) ? (h16)W[(size_t)k * N + n] : (h16)0.0f;
  }
}

__global__ void pad_bias_kernel(const float* __restrict__ b, int N, int Npad,
                                float* __restrict__ out) {
  int n = blockIdx.x * blockDim.x + threadIdx.x;
  if (n < Npad) out[n] = (n < N) ? b[n] : 0.0f;
}

// ------------------------------------------------------------------
// WMMA GEMM, 4x register-blocked in M (64x16 C slab per wave):
//   A  : Mpad(x64) x Kpad f16 row-major (zero-padded cols)
//   Bp : packed [Kpad/32][Npad][32] f16 (zero-padded)
// Per K-step: 1 B fragment (v16h, 2xb128) reused by 4 WMMAs, 8 A b128 loads.
// ------------------------------------------------------------------
__global__ void __launch_bounds__(256) gemm_f16_kernel(
    const h16* __restrict__ A, const h16* __restrict__ Bp,
    const float* __restrict__ biasP, const float* __restrict__ addF,
    float* __restrict__ outF, h16* __restrict__ outH,
    int tilesM4, int tilesN, int Kpad, int relu)
{
  int lane = threadIdx.x & 31;
  int wid  = threadIdx.x >> 5;
  int tile = blockIdx.x * 8 + wid;
  if (tile >= tilesM4 * tilesN) return;     // wave-uniform exit
  int tm = tile / tilesN, tn = tile % tilesN;
  int half_ = lane >> 4, r16 = lane & 15;
  int Npad = tilesN * 16;
  int row0 = tm * 64 + r16;
  int col  = tn * 16 + r16;
  const h16* ap0 = A  + (size_t)row0 * Kpad + half_ * 8;
  const h16* bp  = Bp + (size_t)col * 32    + half_ * 16;
  size_t astep16 = (size_t)16 * Kpad;
  size_t bstep   = (size_t)Npad * 32;
  v8f acc0 = {}, acc1 = {}, acc2 = {}, acc3 = {};
  for (int k0 = 0; k0 < Kpad; k0 += 32) {
    __builtin_prefetch(bp + bstep, 0, 1);   // speculative; OOB dropped by HW
    v16h b = *(const v16h*)bp;              // 16 contiguous halves
    v16h a0 = load_afrag(ap0);
    v16h a1 = load_afrag(ap0 + astep16);
    v16h a2 = load_afrag(ap0 + 2 * astep16);
    v16h a3 = load_afrag(ap0 + 3 * astep16);
    acc0 = __builtin_amdgcn_wmma_f32_16x16x32_f16(false, a0, false, b, (short)0, acc0, false, false);
    acc1 = __builtin_amdgcn_wmma_f32_16x16x32_f16(false, a1, false, b, (short)0, acc1, false, false);
    acc2 = __builtin_amdgcn_wmma_f32_16x16x32_f16(false, a2, false, b, (short)0, acc2, false, false);
    acc3 = __builtin_amdgcn_wmma_f32_16x16x32_f16(false, a3, false, b, (short)0, acc3, false, false);
    ap0 += 32;
    bp  += bstep;
  }
  float bias = biasP[col];
  v8f accs[4] = {acc0, acc1, acc2, acc3};
#pragma unroll
  for (int i = 0; i < 4; ++i) {
#pragma unroll
    for (int r = 0; r < 8; ++r) {
      int m = tm * 64 + i * 16 + r + 8 * half_;   // C layout: VGPR r -> M=r / 8+r
      float v = accs[i][r] + bias;
      if (relu) v = fmaxf(v, 0.0f);
      if (addF) v += addF[(size_t)m * Npad + col];
      if (outF) outF[(size_t)m * Npad + col] = v;
      if (outH) outH[(size_t)m * Npad + col] = (h16)v;
    }
  }
}

// ------------------------------------------------------------------
// Farthest point sampling: one block per segment, LDS dmin + argmax.
// ------------------------------------------------------------------
__global__ void __launch_bounds__(512) fps_kernel(
    const float* __restrict__ pos, int seg_base, int seg_stride, int seg_cnt,
    int m, int* __restrict__ out, int out_base, int out_stride)
{
  __shared__ float dmin[8192];
  __shared__ float redv[512];
  __shared__ int   redi[512];
  __shared__ float lp[3];
  __shared__ int   winner;
  int tid = threadIdx.x;
  int off = seg_base + blockIdx.x * seg_stride;
  int* o  = out + out_base + blockIdx.x * out_stride;
  for (int i = tid; i < seg_cnt; i += 512) dmin[i] = 1e10f;
  if (tid == 0) { o[0] = off; winner = 0; }
  __syncthreads();
  for (int it = 1; it < m; ++it) {
    if (tid < 3) lp[tid] = pos[(size_t)(off + winner) * 3 + tid];
    __syncthreads();
    float x0 = lp[0], y0 = lp[1], z0 = lp[2];
    float bv = -1.0f; int bi = 0;
    for (int i = tid; i < seg_cnt; i += 512) {
      const float* p = pos + (size_t)(off + i) * 3;
      float dx = p[0] - x0, dy = p[1] - y0, dz = p[2] - z0;
      float d  = dx * dx + dy * dy + dz * dz;
      float dm = fminf(dmin[i], d);
      dmin[i] = dm;
      if (dm > bv) { bv = dm; bi = i; }
    }
    redv[tid] = bv; redi[tid] = bi;
    __syncthreads();
    for (int s = 256; s > 0; s >>= 1) {
      if (tid < s) {
        float v2 = redv[tid + s]; int i2 = redi[tid + s];
        if (v2 > redv[tid] || (v2 == redv[tid] && i2 < redi[tid])) {
          redv[tid] = v2; redi[tid] = i2;
        }
      }
      __syncthreads();
    }
    if (tid == 0) { winner = redi[0]; o[it] = off + redi[0]; }
    __syncthreads();
  }
}

// ------------------------------------------------------------------
// Top-K nearest neighbors: one block per query; LDS d2 + K argmins.
// ------------------------------------------------------------------
__global__ void __launch_bounds__(256) knn_kernel(
    const float* __restrict__ pos, const int* __restrict__ qidx, int qbase,
    int seg_off, int seg_cnt, int uniform_seg, int K,
    int* __restrict__ nidx, float* __restrict__ nd2)
{
  __shared__ float d2s[8192];
  __shared__ float redv[256];
  __shared__ int   redi[256];
  int q   = qbase + blockIdx.x;
  int tid = threadIdx.x;
  int qa  = qidx[q];
  int off = seg_off, cnt = seg_cnt;
  if (uniform_seg > 0) { off = (qa / uniform_seg) * uniform_seg; cnt = uniform_seg; }
  float qx = pos[(size_t)qa * 3 + 0];
  float qy = pos[(size_t)qa * 3 + 1];
  float qz = pos[(size_t)qa * 3 + 2];
  for (int i = tid; i < cnt; i += 256) {
    const float* p = pos + (size_t)(off + i) * 3;
    float dx = p[0] - qx, dy = p[1] - qy, dz = p[2] - qz;
    d2s[i] = dx * dx + dy * dy + dz * dz;
  }
  __syncthreads();
  for (int k = 0; k < K; ++k) {
    float bv = 1e30f; int bi = 0;
    for (int i = tid; i < cnt; i += 256) {
      float d = d2s[i];
      if (d < bv) { bv = d; bi = i; }
    }
    redv[tid] = bv; redi[tid] = bi;
    __syncthreads();
    for (int s = 128; s > 0; s >>= 1) {
      if (tid < s) {
        float v2 = redv[tid + s]; int i2 = redi[tid + s];
        if (v2 < redv[tid] || (v2 == redv[tid] && i2 < redi[tid])) {
          redv[tid] = v2; redi[tid] = i2;
        }
      }
      __syncthreads();
    }
    if (tid == 0) {
      nidx[(size_t)q * K + k] = off + redi[0];
      nd2 [(size_t)q * K + k] = redv[0];
      d2s[redi[0]] = 1e30f;
    }
    __syncthreads();
  }
}

// ------------------------------------------------------------------
// Stage-A feature builders (all write zero-padded rows)
// ------------------------------------------------------------------
__global__ void build_sfeat_kernel(const float* __restrict__ pos,
    const h16* __restrict__ xh, const int* __restrict__ gidx,
    h16* __restrict__ sfeat)
{
  size_t row = (size_t)blockIdx.x * blockDim.x + threadIdx.x;
  size_t st  = (size_t)gridDim.x * blockDim.x;
  for (; row < (size_t)8192 * 12; row += st) {
    int q = (int)(row / 12), k = (int)(row % 12);
    int g = gidx[(size_t)q * 32 + k];
    h16* o = sfeat + row * 96;
    for (int d = 0; d < 3; ++d) o[d] = (h16)pos[(size_t)g * 3 + d];
    const h16* xr = xh + (size_t)g * 64;
    for (int c = 0; c < 64; ++c) o[3 + c] = xr[c];
    for (int c = 67; c < 96; ++c) o[c] = (h16)0.0f;
  }
}

__global__ void adapt_newxyz_kernel(const float* __restrict__ wraw,
    const float* __restrict__ pos, const int* __restrict__ gidx,
    float* __restrict__ nxyz)
{
  int q = blockIdx.x * blockDim.x + threadIdx.x;
  if (q >= 8192) return;
  for (int d = 0; d < 3; ++d) {
    float mx = -3.0e38f;
    for (int k = 0; k < 12; ++k)
      mx = fmaxf(mx, wraw[(size_t)(q * 12 + k) * 80 + d]);
    float ex[12]; float se = 0.0f;
    for (int k = 0; k < 12; ++k) {
      ex[k] = expf(wraw[(size_t)(q * 12 + k) * 80 + d] - mx);
      se += ex[k];
    }
    float acc = 0.0f;
    for (int k = 0; k < 12; ++k) {
      int g = gidx[(size_t)q * 32 + k];
      acc += pos[(size_t)g * 3 + d] * (ex[k] / se);
    }
    nxyz[(size_t)q * 3 + d] = acc;
  }
}

__global__ void build_npfeat_kernel(const float* __restrict__ pos,
    const h16* __restrict__ xh, const int* __restrict__ gidx,
    const float* __restrict__ nxyz, h16* __restrict__ npH,
    h16* __restrict__ npT, h16* __restrict__ gxH)
{
  size_t row = (size_t)blockIdx.x * blockDim.x + threadIdx.x;
  size_t st  = (size_t)gridDim.x * blockDim.x;
  for (; row < (size_t)8192 * 32; row += st) {
    int q = (int)(row >> 5), k = (int)(row & 31);
    int g = gidx[row];
    h16* o  = npH + row * 96;
    h16* ot = npT + (size_t)q * 96 * 32 + k;
    const h16* xr = xh + (size_t)g * 64;
    for (int c = 0; c < 96; ++c) {
      float v;
      if (c < 3)       v = pos[(size_t)g * 3 + c] - nxyz[(size_t)q * 3 + c];
      else if (c < 67) v = (float)xr[c - 3];
      else             v = 0.0f;
      h16 hv = (h16)v;
      o[c] = hv;
      ot[(size_t)c * 32] = hv;
    }
    h16* gx = gxH + row * 32;
    for (int d = 0; d < 3; ++d) gx[d] = o[d];
    for (int d = 3; d < 32; ++d) gx[d] = (h16)0.0f;
  }
}

__global__ void rowmax_kernel(const h16* __restrict__ in, int nq, int K, int C,
                              h16* __restrict__ out)
{
  size_t t  = (size_t)blockIdx.x * blockDim.x + threadIdx.x;
  size_t st = (size_t)gridDim.x * blockDim.x;
  size_t tot = (size_t)nq * C;
  for (; t < tot; t += st) {
    int q = (int)(t / C), c = (int)(t % C);
    float mx = -3.0e38f;
    for (int k = 0; k < K; ++k)
      mx = fmaxf(mx, (float)in[((size_t)q * K + k) * C + c]);
    out[t] = (h16)mx;
  }
}

__global__ void pack_wgt_kernel(const h16* __restrict__ wgtH, h16* __restrict__ wgtP)
{
  size_t t  = (size_t)blockIdx.x * blockDim.x + threadIdx.x;
  size_t st = (size_t)gridDim.x * blockDim.x;
  for (; t < (size_t)8192 * 1024; t += st) {
    int q = (int)(t >> 10);
    int r = (int)(t & 1023);
    int w = r >> 5, k = ((r & 8) << 1) | ((r & 16) >> 1) | (r & 7);
    wgtP[t] = wgtH[((size_t)q * 32 + k) * 32 + w];
  }
}

// ------------------------------------------------------------------
// WMMA einsum per point: t[c][w] = sum_k npT[c][k] * weight[k][w]
// One wave per (q, w-tile): B fragment loaded once, 5 WMMAs over c-tiles.
// ------------------------------------------------------------------
__global__ void __launch_bounds__(256) einsum_tw_kernel(
    const h16* __restrict__ npT, const h16* __restrict__ wgtP, h16* __restrict__ tH)
{
  int lane = threadIdx.x & 31;
  int wid  = threadIdx.x >> 5;
  int gw = blockIdx.x * 8 + wid;
  int q  = gw >> 1;
  if (q >= 8192) return;                    // wave-uniform
  int tn = gw & 1;
  int half_ = lane >> 4, r16 = lane & 15;
  int w = tn * 16 + r16;
  v16h b = *(const v16h*)(wgtP + (size_t)q * 1024 + (size_t)w * 32 + half_ * 16);
  const h16* abase = npT + (size_t)q * 96 * 32 + half_ * 8;
#pragma unroll
  for (int tm = 0; tm < 5; ++tm) {
    int c = tm * 16 + r16;                  // c < 80 < 96: pad reads are zeros
    v16h a = load_afrag(abase + (size_t)c * 32);
    v8f acc = {};
    acc = __builtin_amdgcn_wmma_f32_16x16x32_f16(false, a, false, b,
                                                 (short)0, acc, false, false);
#pragma unroll
    for (int r = 0; r < 8; ++r) {
      int cc = tm * 16 + r + 8 * half_;
      if (cc < 67) tH[(size_t)q * 2144 + (size_t)cc * 32 + w] = (h16)acc[r];
    }
  }
}

// ------------------------------------------------------------------
// SA-module helpers
// ------------------------------------------------------------------
__global__ void gather3_kernel(const float* __restrict__ pos,
    const int* __restrict__ qi, int m, float* __restrict__ out)
{
  size_t t  = (size_t)blockIdx.x * blockDim.x + threadIdx.x;
  size_t st = (size_t)gridDim.x * blockDim.x;
  for (; t < (size_t)m * 3; t += st)
    out[t] = pos[(size_t)qi[t / 3] * 3 + (t % 3)];
}

__global__ void build_safeat_kernel(const float* __restrict__ pos,
    const h16* __restrict__ xh, const int* __restrict__ qi,
    const int* __restrict__ nidx, int nq, int K, int C, int lda,
    h16* __restrict__ feat)
{
  size_t row = (size_t)blockIdx.x * blockDim.x + threadIdx.x;
  size_t st  = (size_t)gridDim.x * blockDim.x;
  for (; row < (size_t)nq * K; row += st) {
    int q  = (int)(row / K);
    int n  = nidx[row];
    int qa = qi[q];
    h16* o = feat + row * lda;
    const h16* xr = xh + (size_t)n * C;
    for (int c = 0; c < C; ++c) o[c] = xr[c];
    for (int d = 0; d < 3; ++d)
      o[C + d] = (h16)(pos[(size_t)n * 3 + d] - pos[(size_t)qa * 3 + d]);
    for (int c = C + 3; c < lda; ++c) o[c] = (h16)0.0f;
  }
}

__global__ void maxpool_valid_kernel(const h16* __restrict__ h,
    const float* __restrict__ nd2, float r2, int nq, int K, int C,
    h16* __restrict__ outH)
{
  size_t t  = (size_t)blockIdx.x * blockDim.x + threadIdx.x;
  size_t st = (size_t)gridDim.x * blockDim.x;
  size_t tot = (size_t)nq * C;
  for (; t < tot; t += st) {
    int q = (int)(t / C), c = (int)(t % C);
    float mx = -1.0e9f;
    for (int k = 0; k < K; ++k)
      if (nd2[(size_t)q * K + k] <= r2)
        mx = fmaxf(mx, (float)h[((size_t)q * K + k) * C + c]);
    outH[t] = (h16)mx;
  }
}

__global__ void concat_xp_kernel(const h16* __restrict__ x3h,
    const float* __restrict__ p3, h16* __restrict__ out)
{
  size_t t  = (size_t)blockIdx.x * blockDim.x + threadIdx.x;
  size_t st = (size_t)gridDim.x * blockDim.x;
  for (; t < (size_t)512 * 288; t += st) {
    int r = (int)(t / 288), c = (int)(t % 288);
    h16 v;
    if (c < 256)      v = x3h[(size_t)r * 256 + c];
    else if (c < 259) v = (h16)p3[(size_t)r * 3 + (c - 256)];
    else              v = (h16)0.0f;
    out[t] = v;
  }
}

__global__ void segmax_kernel(const h16* __restrict__ h, int roff, int rcnt,
                              int C, h16* __restrict__ out)
{
  int c = blockIdx.x * blockDim.x + threadIdx.x;
  if (c >= C) return;
  float mx = -3.0e38f;
  for (int r = 0; r < rcnt; ++r)
    mx = fmaxf(mx, (float)h[(size_t)(roff + r) * C + c]);
  out[c] = (h16)mx;
}

__global__ void logsoftmax_kernel(const float* __restrict__ logits,
                                  float* __restrict__ out)
{
  int i = threadIdx.x;
  if (i < 2) {
    float a = logits[i * 16 + 0], b = logits[i * 16 + 1];
    float mx = fmaxf(a, b);
    float lse = mx + logf(expf(a - mx) + expf(b - mx));
    out[i * 2 + 0] = a - lse;
    out[i * 2 + 1] = b - lse;
  }
}

// ------------------------------------------------------------------
// Host orchestration
// ------------------------------------------------------------------
extern "C" void kernel_launch(void* const* d_in, const int* in_sizes, int n_in,
                              void* d_out, int out_size, void* d_ws, size_t ws_size,
                              hipStream_t stream)
{
  (void)in_sizes; (void)n_in; (void)out_size; (void)ws_size;
  const float* pos = (const float*)d_in[1];

  char* base = (char*)d_ws;
  size_t off = 0;
  auto alloc = [&](size_t bytes) -> void* {
    void* p = base + off;
    off += (bytes + 255) & ~(size_t)255;
    return p;
  };
  auto gs = [](size_t n) -> dim3 {
    size_t bks = (n + 255) / 256;
    if (bks > 32768) bks = 32768;
    return dim3((unsigned)bks);
  };
  auto toh = [&](const void* src, size_t n) -> h16* {
    h16* d = (h16*)alloc(sizeof(h16) * n);
    f32_to_f16_kernel<<<gs(n), 256, 0, stream>>>((const float*)src, d, n);
    return d;
  };
  auto packB = [&](const void* W, int K, int N, int Kpad, int Npad) -> h16* {
    size_t n = (size_t)Kpad * Npad;
    h16* d = (h16*)alloc(sizeof(h16) * n);
    pack_b_kernel<<<gs(n), 256, 0, stream>>>((const float*)W, K, N, Npad, d, n);
    return d;
  };
  auto padb = [&](const void* b, int N, int Npad) -> float* {
    float* d = (float*)alloc(4 * (size_t)Npad);
    pad_bias_kernel<<<gs(Npad), 256, 0, stream>>>((const float*)b, N, Npad, d);
    return d;
  };
  auto gemm = [&](const h16* A, const h16* Bp, const float* bP, const float* add,
                  float* oF, h16* oH, int M, int Npad, int Kpad, int relu) {
    int tilesM4 = (M + 63) / 64;
    int tilesN  = Npad / 16;
    int blocks  = (tilesM4 * tilesN + 7) / 8;
    gemm_f16_kernel<<<dim3(blocks), dim3(256), 0, stream>>>(
        A, Bp, bP, add, oF, oH, tilesM4, tilesN, Kpad, relu);
  };

  // fp16 activations
  h16* xh = toh(d_in[0], (size_t)131072 * 64);
  // packed weights (Kpad x Npad) + padded biases
  h16* W1p  = packB(d_in[2], 67, 32, 96, 32);     float* b1p  = padb(d_in[3], 32, 32);
  h16* W2p  = packB(d_in[4], 32, 67, 32, 80);     float* b2p  = padb(d_in[5], 67, 80);
  h16* Wskp = packB(d_in[6], 67, 64, 96, 64);     float* bskp = padb(d_in[7], 64, 64);
  h16* Wwp  = packB(d_in[8], 3, 32, 32, 32);      float* bwp  = padb(d_in[9], 32, 32);
  h16* Wcp  = packB(d_in[10], 2144, 64, 2144, 64);float* bcp  = padb(d_in[11], 64, 64);
  h16* s1w0 = packB(d_in[12], 67, 64, 96, 64);    float* s1b0 = padb(d_in[15], 64, 64);
  h16* s1w1 = packB(d_in[13], 64, 64, 64, 64);    float* s1b1 = padb(d_in[16], 64, 64);
  h16* s1w2 = packB(d_in[14], 64, 96, 64, 96);    float* s1b2 = padb(d_in[17], 96, 96);
  h16* s2w0 = packB(d_in[18], 99, 96, 128, 96);   float* s2b0 = padb(d_in[21], 96, 96);
  h16* s2w1 = packB(d_in[19], 96, 96, 96, 96);    float* s2b1 = padb(d_in[22], 96, 96);
  h16* s2w2 = packB(d_in[20], 96, 128, 96, 128);  float* s2b2 = padb(d_in[23], 128, 128);
  h16* s3w0 = packB(d_in[24], 131, 128, 160, 128);float* s3b0 = padb(d_in[27], 128, 128);
  h16* s3w1 = packB(d_in[25], 128, 128, 128, 128);float* s3b1 = padb(d_in[28], 128, 128);
  h16* s3w2 = packB(d_in[26], 128, 256, 128, 256);float* s3b2 = padb(d_in[29], 256, 256);
  h16* s4w0 = packB(d_in[30], 259, 256, 288, 256);float* s4b0 = padb(d_in[33], 256, 256);
  h16* s4w1 = packB(d_in[31], 256, 512, 256, 512);float* s4b1 = padb(d_in[34], 512, 512);
  h16* s4w2 = packB(d_in[32], 512, 1024, 512, 1024);float* s4b2 = padb(d_in[35], 1024, 1024);
  h16* hw0  = packB(d_in[36], 1024, 512, 1024, 512);float* hb0 = padb(d_in[40], 512, 512);
  h16* hw1  = packB(d_in[37], 512, 256, 512, 256);float* hb1  = padb(d_in[41], 256, 256);
  h16* hw2  = packB(d_in[38], 256, 128, 256, 128);float* hb2  = padb(d_in[42], 128, 128);
  h16* hw3  = packB(d_in[39], 128, 2, 128, 16);   float* hb3  = padb(d_in[43], 2, 16);

  // workspace buffers (padded layouts; small matrices padded to 64 rows)
  int*   qidxA  = (int*)alloc(4 * (size_t)8192);
  int*   gidx   = (int*)alloc(4 * (size_t)8192 * 32);
  float* gd2    = (float*)alloc(4 * (size_t)8192 * 32);
  h16*   sfeatH = (h16*)alloc(2 * (size_t)98304 * 96);
  h16*   h1H    = (h16*)alloc(2 * (size_t)98304 * 32);
  float* wraw   = (float*)alloc(4 * (size_t)98304 * 80);
  float* pos0   = (float*)alloc(4 * (size_t)8192 * 3);
  h16*   npH    = (h16*)alloc(2 * (size_t)262144 * 96);
  h16*   npT    = (h16*)alloc(2 * (size_t)8192 * 96 * 32);
  h16*   gxH    = (h16*)alloc(2 * (size_t)262144 * 32);
  h16*   npmaxH = (h16*)alloc(2 * (size_t)8192 * 96);
  float* skipF  = (float*)alloc(4 * (size_t)8192 * 64);
  h16*   wgtH   = (h16*)alloc(2 * (size_t)262144 * 32);
  h16*   wgtP   = (h16*)alloc(2 * (size_t)8192 * 1024);
  h16*   tH     = (h16*)alloc(2 * (size_t)8192 * 2144);
  h16*   x0H    = (h16*)alloc(2 * (size_t)8192 * 64);
  int*   qi1    = (int*)alloc(4 * (size_t)4096);
  float* pos1   = (float*)alloc(4 * (size_t)4096 * 3);
  int*   qi2    = (int*)alloc(4 * (size_t)2048);
  float* pos2   = (float*)alloc(4 * (size_t)2048 * 3);
  int*   qi3    = (int*)alloc(4 * (size_t)512);
  float* pos3   = (float*)alloc(4 * (size_t)512 * 3);
  int*   nidx   = (int*)alloc(4 * (size_t)262144);
  float* nd2    = (float*)alloc(4 * (size_t)262144);
  h16*   featH  = (h16*)alloc(2 * (size_t)262144 * 96);
  h16*   bufA   = (h16*)alloc(2 * (size_t)262144 * 96);
  h16*   bufB   = (h16*)alloc(2 * (size_t)262144 * 96);
  h16*   x1H    = (h16*)alloc(2 * (size_t)4096 * 96);
  h16*   x2H    = (h16*)alloc(2 * (size_t)2048 * 128);
  h16*   x3H    = (h16*)alloc(2 * (size_t)512 * 256);
  h16*   cat3H  = (h16*)alloc(2 * (size_t)512 * 288);
  h16*   gH     = (h16*)alloc(2 * (size_t)64 * 1024);
  float* logitsF = (float*)alloc(4 * (size_t)64 * 16);

  // ---------------- Stage A ----------------
  fps_kernel<<<16, 512, 0, stream>>>(pos, 0, 8192, 8192, 512, qidxA, 0, 512);
  knn_kernel<<<8192, 256, 0, stream>>>(pos, qidxA, 0, 0, 0, 8192, 32, gidx, gd2);
  build_sfeat_kernel<<<gs(98304), 256, 0, stream>>>(pos, xh, gidx, sfeatH);
  gemm(sfeatH, W1p, b1p, nullptr, nullptr, h1H, 98304, 32, 96, 1);
  gemm(h1H, W2p, b2p, nullptr, wraw, nullptr, 98304, 80, 32, 0);
  adapt_newxyz_kernel<<<gs(8192), 256, 0, stream>>>(wraw, pos, gidx, pos0);
  build_npfeat_kernel<<<gs(262144), 256, 0, stream>>>(pos, xh, gidx, pos0, npH, npT, gxH);
  rowmax_kernel<<<gs((size_t)8192 * 96), 256, 0, stream>>>(npH, 8192, 32, 96, npmaxH);
  gemm(npmaxH, Wskp, bskp, nullptr, skipF, nullptr, 8192, 64, 96, 1);
  gemm(gxH, Wwp, bwp, nullptr, nullptr, wgtH, 262144, 32, 32, 1);
  pack_wgt_kernel<<<gs((size_t)8192 * 1024), 256, 0, stream>>>(wgtH, wgtP);
  einsum_tw_kernel<<<dim3((8192 * 2 + 7) / 8), 256, 0, stream>>>(npT, wgtP, tH);
  gemm(tH, Wcp, bcp, skipF, nullptr, x0H, 8192, 64, 2144, 1);  // relu(t@W+b)+skip

  // ---------------- SA1: [512,7680] -> [256,3840], r=0.2 ----------------
  fps_kernel<<<1, 512, 0, stream>>>(pos0, 0, 0, 512, 256, qi1, 0, 0);
  fps_kernel<<<1, 512, 0, stream>>>(pos0, 512, 0, 7680, 3840, qi1, 256, 0);
  gather3_kernel<<<gs(4096 * 3), 256, 0, stream>>>(pos0, qi1, 4096, pos1);
  knn_kernel<<<256, 256, 0, stream>>>(pos0, qi1, 0, 0, 512, 0, 64, nidx, nd2);
  knn_kernel<<<3840, 256, 0, stream>>>(pos0, qi1, 256, 512, 7680, 0, 64, nidx, nd2);
  build_safeat_kernel<<<gs(262144), 256, 0, stream>>>(pos0, x0H, qi1, nidx, 4096, 64, 64, 96, featH);
  gemm(featH, s1w0, s1b0, nullptr, nullptr, bufA, 262144, 64, 96, 1);
  gemm(bufA, s1w1, s1b1, nullptr, nullptr, bufB, 262144, 64, 64, 1);
  gemm(bufB, s1w2, s1b2, nullptr, nullptr, bufA, 262144, 96, 64, 1);
  maxpool_valid_kernel<<<gs((size_t)4096 * 96), 256, 0, stream>>>(bufA, nd2, 0.04f, 4096, 64, 96, x1H);

  // ---------------- SA1a: [256,3840] -> [128,1920], r=0.2 ----------------
  fps_kernel<<<1, 512, 0, stream>>>(pos1, 0, 0, 256, 128, qi2, 0, 0);
  fps_kernel<<<1, 512, 0, stream>>>(pos1, 256, 0, 3840, 1920, qi2, 128, 0);
  gather3_kernel<<<gs(2048 * 3), 256, 0, stream>>>(pos1, qi2, 2048, pos2);
  knn_kernel<<<128, 256, 0, stream>>>(pos1, qi2, 0, 0, 256, 0, 64, nidx, nd2);
  knn_kernel<<<1920, 256, 0, stream>>>(pos1, qi2, 128, 256, 3840, 0, 64, nidx, nd2);
  build_safeat_kernel<<<gs(131072), 256, 0, stream>>>(pos1, x1H, qi2, nidx, 2048, 64, 96, 128, featH);
  gemm(featH, s2w0, s2b0, nullptr, nullptr, bufA, 131072, 96, 128, 1);
  gemm(bufA, s2w1, s2b1, nullptr, nullptr, bufB, 131072, 96, 96, 1);
  gemm(bufB, s2w2, s2b2, nullptr, nullptr, bufA, 131072, 128, 96, 1);
  maxpool_valid_kernel<<<gs((size_t)2048 * 128), 256, 0, stream>>>(bufA, nd2, 0.04f, 2048, 64, 128, x2H);

  // ---------------- SA2: [128,1920] -> [32,480], r=0.4 ----------------
  fps_kernel<<<1, 512, 0, stream>>>(pos2, 0, 0, 128, 32, qi3, 0, 0);
  fps_kernel<<<1, 512, 0, stream>>>(pos2, 128, 0, 1920, 480, qi3, 32, 0);
  gather3_kernel<<<gs(512 * 3), 256, 0, stream>>>(pos2, qi3, 512, pos3);
  knn_kernel<<<32, 256, 0, stream>>>(pos2, qi3, 0, 0, 128, 0, 64, nidx, nd2);
  knn_kernel<<<480, 256, 0, stream>>>(pos2, qi3, 32, 128, 1920, 0, 64, nidx, nd2);
  build_safeat_kernel<<<gs(32768), 256, 0, stream>>>(pos2, x2H, qi3, nidx, 512, 64, 128, 160, featH);
  gemm(featH, s3w0, s3b0, nullptr, nullptr, bufA, 32768, 128, 160, 1);
  gemm(bufA, s3w1, s3b1, nullptr, nullptr, bufB, 32768, 128, 128, 1);
  gemm(bufB, s3w2, s3b2, nullptr, nullptr, bufA, 32768, 256, 128, 1);
  maxpool_valid_kernel<<<gs((size_t)512 * 256), 256, 0, stream>>>(bufA, nd2, 0.16f, 512, 64, 256, x3H);

  // ---------------- SA3 MLP + segment pooling ----------------
  concat_xp_kernel<<<gs((size_t)512 * 288), 256, 0, stream>>>(x3H, pos3, cat3H);
  gemm(cat3H, s4w0, s4b0, nullptr, nullptr, bufA, 512, 256, 288, 1);
  gemm(bufA, s4w1, s4b1, nullptr, nullptr, bufB, 512, 512, 256, 1);
  gemm(bufB, s4w2, s4b2, nullptr, nullptr, bufA, 512, 1024, 512, 1);
  segmax_kernel<<<4, 256, 0, stream>>>(bufA, 0, 32, 1024, gH);
  segmax_kernel<<<4, 256, 0, stream>>>(bufA, 32, 480, 1024, gH + 1024);

  // ---------------- Head + log_softmax ----------------
  gemm(gH, hw0, hb0, nullptr, nullptr, bufB, 2, 512, 1024, 1);
  gemm(bufB, hw1, hb1, nullptr, nullptr, bufA, 2, 256, 512, 1);
  gemm(bufA, hw2, hb2, nullptr, nullptr, bufB, 2, 128, 256, 1);
  gemm(bufB, hw3, hb3, nullptr, logitsF, nullptr, 2, 16, 128, 0);
  logsoftmax_kernel<<<1, 32, 0, stream>>>(logitsF, (float*)d_out);
}
